// GAT_12249246728929
// MI455X (gfx1250) — matrix operand order repaired and checked by reading, not compiled, not run on hardware
//
#include <hip/hip_runtime.h>
#include <hip/hip_bf16.h>

// ---------------- problem constants (match reference) ----------------
#define N_NODES 50000
#define N_EDGES 800000
#define ETOT    (N_NODES + N_EDGES)   // 850000 (self loops appended)
#define HC      128
#define NHEAD   4
#define CH      32
#define NGRAPH  500
#define NCLS    6
#define BN_EPS  1e-5f
#define NEG_SLOPE 0.2f

#define WT_STRIDE 136   // 128 + 8 f16 pad -> 272B row stride, bank-conflict-free

typedef __attribute__((ext_vector_type(16))) _Float16 v16h;
typedef __attribute__((ext_vector_type(8)))  float    v8f;
typedef int async_v4i __attribute__((vector_size(16)));   // pointee type of async-LDS builtins

#if defined(__has_builtin)
#if __has_builtin(__builtin_amdgcn_global_load_async_to_lds_b128) && \
    __has_builtin(__builtin_amdgcn_s_wait_asynccnt)
#define USE_ASYNC_LDS 1
#endif
#endif

// ---------------- helpers ----------------
__device__ __forceinline__ float atomAddF(float* p, float v) {
  return __hip_atomic_fetch_add(p, v, __ATOMIC_RELAXED, __HIP_MEMORY_SCOPE_AGENT);
}

// monotonic float<->uint key for atomic float-max via u32 max
__device__ __forceinline__ unsigned fkey(float f) {
  int i = __float_as_int(f);
  return (i >= 0) ? ((unsigned)i | 0x80000000u) : ~(unsigned)i;
}
__device__ __forceinline__ float funkey(unsigned u) {
  return (u & 0x80000000u) ? __int_as_float((int)(u & 0x7FFFFFFFu))
                           : __int_as_float((int)~u);
}

// ---------------- 0) weight prep: Wt[col][k] = (f16) W[k][col] ----------------
__global__ __launch_bounds__(256)
void wt_prep_k(const float* __restrict__ W, _Float16* __restrict__ Wt)
{
  const int i = blockIdx.x * blockDim.x + threadIdx.x;
  if (i >= HC * HC) return;
  const int c = i >> 7;
  const int k = i & 127;
  Wt[(size_t)c * HC + k] = (_Float16)W[(size_t)k * HC + c];
}

// ---------------- 1) projection GEMM: D[M,128] = A[M,128] @ W[128,128] ----------------
// One wave per 16x16 output tile; block of 8 waves covers one 16-row stripe
// across all 8 column tiles. Wt (f16, col-major) is staged once per block into
// padded LDS via the gfx1250 async-to-LDS path, then each K-step is one
// v_wmma_f32_16x16x32_f16 with B fragments read as contiguous ds_load_b128.
// Fragment packing (wave32; A/B use a contiguous k order, not the ISA table order):
//   A (16x32 f16): row = lane%16; elem j -> K = k0+16h+j
//   B (32x16 f16): col = lane%16; elem j -> K = k0+16h+j
//   C/D (16x16 f32): vgpr i -> row = i+8h, col = lane%16
__global__ __launch_bounds__(256)
void gemm128_wmma_k(const float* __restrict__ A, const _Float16* __restrict__ Wt,
                    float* __restrict__ D, int M)
{
  __shared__ _Float16 ldsW[WT_STRIDE * HC];   // 34816 B

  // ---- cooperative stage of Wt into LDS (32KB payload) ----
  for (int i = threadIdx.x; i < HC * 16; i += 256) {   // 2048 x 16B segments
    const int c = i >> 4;          // column 0..127
    const int s = i & 15;          // 8-f16 segment 0..15
    const _Float16* g = Wt + (size_t)c * HC + s * 8;
    _Float16* l = &ldsW[c * WT_STRIDE + s * 8];
#if defined(USE_ASYNC_LDS)
    __builtin_amdgcn_global_load_async_to_lds_b128(
        (__attribute__((address_space(1))) async_v4i*)g,
        (__attribute__((address_space(3))) async_v4i*)l, 0, 0);
#else
    *(float4*)l = *(const float4*)g;
#endif
  }
#if defined(USE_ASYNC_LDS)
  __builtin_amdgcn_s_wait_asynccnt(0);
#endif
  __syncthreads();

  const int wave = (int)((blockIdx.x * blockDim.x + threadIdx.x) >> 5);
  const int lane = threadIdx.x & 31;
  const int tm = wave >> 3;          // row tile
  const int tn = wave & 7;           // col tile (128/16 = 8)
  if (tm * 16 >= M) return;
  const int half = lane >> 4;
  const int lr   = lane & 15;
  const float* Arow = A + (size_t)(tm * 16 + lr) * HC;
  const int col = tn * 16 + lr;
  const _Float16* Bcol = &ldsW[col * WT_STRIDE];

  v8f c = {};
  #pragma unroll
  for (int k0 = 0; k0 < 128; k0 += 32) {
    v16h a, b;
    #pragma unroll
    for (int j = 0; j < 8; ++j) {
      a[j]     = (_Float16)Arow[k0 + 16 * half + j];
      a[j + 8] = (_Float16)Arow[k0 + 16 * half + 8 + j];
    }
    const _Float16* bp = Bcol + k0 + 16 * half;   // 16 contiguous f16 (32B)
    #pragma unroll
    for (int j = 0; j < 16; ++j)
      b[j] = bp[j];
    c = __builtin_amdgcn_wmma_f32_16x16x32_f16(false, a, false, b,
                                               (short)0, c, false, false);
  }
  #pragma unroll
  for (int i = 0; i < 8; ++i)
    D[(size_t)(tm * 16 + i + 8 * half) * HC + col] = c[i];
}

// ---------------- 2) attention logits: al/ar[n,h] = <h[n,h,:], a_src/dst[h,:]> ----------------
__global__ __launch_bounds__(256)
void attn_proj_k(const float* __restrict__ h, const float* __restrict__ asv,
                 const float* __restrict__ adv,
                 float* __restrict__ al, float* __restrict__ ar)
{
  const int node = blockIdx.x * 8 + (threadIdx.x >> 5);
  const int lane = threadIdx.x & 31;
  if (node >= N_NODES) return;
  const float* hr = h + (size_t)node * HC;
  #pragma unroll
  for (int hd = 0; hd < NHEAD; ++hd) {
    const float v = hr[hd * CH + lane];
    float s = v * asv[hd * CH + lane];
    float d = v * adv[hd * CH + lane];
    #pragma unroll
    for (int off = 16; off; off >>= 1) {
      s += __shfl_xor(s, off, 32);
      d += __shfl_xor(d, off, 32);
    }
    if (lane == 0) {
      al[(size_t)node * NHEAD + hd] = s;
      ar[(size_t)node * NHEAD + hd] = d;
    }
  }
}

// ---------------- 3) per-dst max of leaky-relu scores (atomic u32 max on ordered keys) ----------------
__global__ __launch_bounds__(256)
void edge_max_k(const int* __restrict__ ei, const float* __restrict__ al,
                const float* __restrict__ ar, unsigned* __restrict__ mkey)
{
  const int e = blockIdx.x * blockDim.x + threadIdx.x;
  if (e >= ETOT) return;
  int s, d;
  if (e < N_EDGES) { s = ei[e]; d = ei[N_EDGES + e]; } else { s = d = e - N_EDGES; }
  const float4 as = *(const float4*)(al + (size_t)s * NHEAD);
  const float4 ad = *(const float4*)(ar + (size_t)d * NHEAD);
  float sc[4] = {as.x + ad.x, as.y + ad.y, as.z + ad.z, as.w + ad.w};
  #pragma unroll
  for (int hd = 0; hd < NHEAD; ++hd) {
    float v = sc[hd];
    v = v > 0.f ? v : NEG_SLOPE * v;
    atomicMax(&mkey[(size_t)d * NHEAD + hd], fkey(v));
  }
}

// ---------------- 4) p = exp(score - max); segment-sum of p per dst ----------------
__global__ __launch_bounds__(256)
void edge_exp_k(const int* __restrict__ ei, const float* __restrict__ al,
                const float* __restrict__ ar, const unsigned* __restrict__ mkey,
                float* __restrict__ ssum, float* __restrict__ pbuf)
{
  const int e = blockIdx.x * blockDim.x + threadIdx.x;
  if (e >= ETOT) return;
  int s, d;
  if (e < N_EDGES) { s = ei[e]; d = ei[N_EDGES + e]; } else { s = d = e - N_EDGES; }
  const float4 as = *(const float4*)(al + (size_t)s * NHEAD);
  const float4 ad = *(const float4*)(ar + (size_t)d * NHEAD);
  float sc[4] = {as.x + ad.x, as.y + ad.y, as.z + ad.z, as.w + ad.w};
  float4 pv;
  float* pp = (float*)&pv;
  #pragma unroll
  for (int hd = 0; hd < NHEAD; ++hd) {
    float v = sc[hd];
    v = v > 0.f ? v : NEG_SLOPE * v;
    const float m = funkey(mkey[(size_t)d * NHEAD + hd]);
    const float p = expf(v - m);
    pp[hd] = p;
    atomAddF(&ssum[(size_t)d * NHEAD + hd], p);
  }
  *(float4*)(pbuf + (size_t)e * NHEAD) = pv;
}

// ---------------- 5) aggregation: agg[dst] += alpha * h[src]  (one wave per edge) ----------------
__global__ __launch_bounds__(256)
void edge_agg_k(const int* __restrict__ ei, const float* __restrict__ h,
                const float* __restrict__ pbuf, const float* __restrict__ ssum,
                float* __restrict__ agg)
{
  const int wave = (int)((blockIdx.x * blockDim.x + threadIdx.x) >> 5);
  const int lane = threadIdx.x & 31;
  if (wave >= ETOT) return;
  int s, d;
  if (wave < N_EDGES) { s = ei[wave]; d = ei[N_EDGES + wave]; } else { s = d = wave - N_EDGES; }
  const float* hs = h + (size_t)s * HC;
  float* ag = agg + (size_t)d * HC;
  __builtin_prefetch(hs + lane, 0, 1);   // pull gathered row toward the WGP
  #pragma unroll
  for (int hd = 0; hd < NHEAD; ++hd) {
    const float alpha = pbuf[(size_t)wave * NHEAD + hd] /
                        ssum[(size_t)d * NHEAD + hd];
    atomAddF(&ag[hd * CH + lane], alpha * hs[hd * CH + lane]);
  }
}

// ---------------- 6) BN stats: per-column sum & sumsq of (agg + bias) ----------------
__global__ __launch_bounds__(128)
void bn_stats_k(const float* __restrict__ x, const float* __restrict__ bias,
                float* __restrict__ stats)
{
  const int c  = threadIdx.x;      // 128 columns
  const int r0 = blockIdx.x * 256; // 256 rows per block
  const float b = bias[c];
  float s = 0.f, q = 0.f;
  for (int r = 0; r < 256; ++r) {
    const int row = r0 + r;
    if (row >= N_NODES) break;
    const float v = x[(size_t)row * HC + c] + b;
    s += v; q += v * v;
  }
  atomAddF(&stats[c], s);
  atomAddF(&stats[HC + c], q);
}

// ---------------- 7) BN apply + ELU ----------------
__global__ __launch_bounds__(256)
void bn_apply_k(const float* __restrict__ x, const float* __restrict__ bias,
                const float* __restrict__ stats, const float* __restrict__ g,
                const float* __restrict__ be, float* __restrict__ y)
{
  const size_t i = (size_t)blockIdx.x * blockDim.x + threadIdx.x;
  if (i >= (size_t)N_NODES * HC) return;
  const int c = (int)(i & (HC - 1));
  const float inv_n = 1.0f / (float)N_NODES;
  const float mu  = stats[c] * inv_n;
  const float var = stats[HC + c] * inv_n - mu * mu;
  const float v = x[i] + bias[c];
  const float z = g[c] * (v - mu) * rsqrtf(var + BN_EPS) + be[c];
  y[i] = z > 0.f ? z : expm1f(z);
}

// ---------------- 8) pooling (sum) + counts ----------------
__global__ __launch_bounds__(256)
void pool_k(const float* __restrict__ hn, const int* __restrict__ batch,
            float* __restrict__ pooled)
{
  const size_t i = (size_t)blockIdx.x * blockDim.x + threadIdx.x;
  if (i >= (size_t)N_NODES * HC) return;
  const int n = (int)(i >> 7);
  const int c = (int)(i & (HC - 1));
  const int gi = batch[n];
  atomAddF(&pooled[(size_t)gi * HC + c], hn[i]);
}

__global__ __launch_bounds__(256)
void cnt_k(const int* __restrict__ batch, float* __restrict__ cnt)
{
  const int n = blockIdx.x * blockDim.x + threadIdx.x;
  if (n >= N_NODES) return;
  atomAddF(&cnt[batch[n]], 1.0f);
}

// ---------------- 9) final FC: out[g,k] = (pooled[g]/cnt) @ fcW + fcb ----------------
__global__ __launch_bounds__(256)
void fc_k(const float* __restrict__ pooled, const float* __restrict__ cnt,
          const float* __restrict__ W, const float* __restrict__ bb,
          float* __restrict__ out)
{
  const int i = blockIdx.x * blockDim.x + threadIdx.x;
  if (i >= NGRAPH * NCLS) return;
  const int gi = i / NCLS;
  const int k  = i % NCLS;
  const float inv = 1.f / fmaxf(cnt[gi], 1.f);
  float acc = 0.f;
  #pragma unroll 4
  for (int c = 0; c < HC; ++c)
    acc += pooled[(size_t)gi * HC + c] * W[c * NCLS + k];
  out[i] = acc * inv + bb[k];
}

// ---------------- host-side layer driver ----------------
static void run_gat_layer(const float* xin, const int* ei,
                          const float* W, const float* asv, const float* adv,
                          const float* bias, const float* gamma, const float* beta,
                          _Float16* Wt, float* h, float* al, float* ar,
                          unsigned* mkey, float* ssum, float* pbuf, float* agg,
                          float* stats, float* yout, hipStream_t stream)
{
  (void)hipMemsetAsync(mkey, 0, (size_t)N_NODES * NHEAD * sizeof(unsigned), stream);
  (void)hipMemsetAsync(ssum, 0, (size_t)N_NODES * NHEAD * sizeof(float), stream);
  (void)hipMemsetAsync(agg,  0, (size_t)N_NODES * HC * sizeof(float), stream);
  (void)hipMemsetAsync(stats, 0, 2 * HC * sizeof(float), stream);

  wt_prep_k<<<(HC * HC + 255) / 256, 256, 0, stream>>>(W, Wt);
  {
    const int waves = (N_NODES / 16) * 8;          // 25000 tiles
    const int thr   = waves * 32;
    gemm128_wmma_k<<<(thr + 255) / 256, 256, 0, stream>>>(xin, Wt, h, N_NODES);
  }
  attn_proj_k<<<(N_NODES + 7) / 8, 256, 0, stream>>>(h, asv, adv, al, ar);
  edge_max_k<<<(ETOT + 255) / 256, 256, 0, stream>>>(ei, al, ar, mkey);
  edge_exp_k<<<(ETOT + 255) / 256, 256, 0, stream>>>(ei, al, ar, mkey, ssum, pbuf);
  {
    const long long thr = (long long)ETOT * 32;
    edge_agg_k<<<(unsigned)((thr + 255) / 256), 256, 0, stream>>>(ei, h, pbuf, ssum, agg);
  }
  bn_stats_k<<<(N_NODES + 255) / 256, 128, 0, stream>>>(agg, bias, stats);
  {
    const long long t = (long long)N_NODES * HC;
    bn_apply_k<<<(unsigned)((t + 255) / 256), 256, 0, stream>>>(agg, bias, stats,
                                                                gamma, beta, yout);
  }
}

extern "C" void kernel_launch(void* const* d_in, const int* in_sizes, int n_in,
                              void* d_out, int out_size, void* d_ws, size_t ws_size,
                              hipStream_t stream)
{
  const float* x     = (const float*)d_in[0];
  const int*   ei    = (const int*)d_in[1];   // [2, E] row-major: src then dst
  const int*   batch = (const int*)d_in[2];
  const float* W1  = (const float*)d_in[3];
  const float* as1 = (const float*)d_in[4];
  const float* ad1 = (const float*)d_in[5];
  const float* b1  = (const float*)d_in[6];
  const float* g1  = (const float*)d_in[7];
  const float* be1 = (const float*)d_in[8];
  const float* W2  = (const float*)d_in[9];
  const float* as2 = (const float*)d_in[10];
  const float* ad2 = (const float*)d_in[11];
  const float* b2  = (const float*)d_in[12];
  const float* g2  = (const float*)d_in[13];
  const float* be2 = (const float*)d_in[14];
  const float* fcW = (const float*)d_in[15];
  const float* fcb = (const float*)d_in[16];
  float* out = (float*)d_out;

  // workspace carve-up (256B aligned slices)
  char* w = (char*)d_ws;
  auto alloc = [&](size_t bytes) -> char* {
    char* p = w;
    w += (bytes + 255) & ~(size_t)255;
    return p;
  };
  float*    h      = (float*)   alloc((size_t)N_NODES * HC * sizeof(float));
  float*    hn     = (float*)   alloc((size_t)N_NODES * HC * sizeof(float));
  float*    agg    = (float*)   alloc((size_t)N_NODES * HC * sizeof(float));
  float*    al     = (float*)   alloc((size_t)N_NODES * NHEAD * sizeof(float));
  float*    ar     = (float*)   alloc((size_t)N_NODES * NHEAD * sizeof(float));
  unsigned* mkey   = (unsigned*)alloc((size_t)N_NODES * NHEAD * sizeof(unsigned));
  float*    ssum   = (float*)   alloc((size_t)N_NODES * NHEAD * sizeof(float));
  float*    pbuf   = (float*)   alloc((size_t)ETOT * NHEAD * sizeof(float));
  float*    stats  = (float*)   alloc(2 * HC * sizeof(float));
  float*    pooled = (float*)   alloc((size_t)NGRAPH * HC * sizeof(float));
  float*    cnt    = (float*)   alloc((size_t)NGRAPH * sizeof(float));
  _Float16* Wt     = (_Float16*)alloc((size_t)HC * HC * sizeof(_Float16));

  // ----- layer 1: x -> hn -----
  run_gat_layer(x, ei, W1, as1, ad1, b1, g1, be1,
                Wt, h, al, ar, mkey, ssum, pbuf, agg, stats, hn, stream);
  // ----- layer 2: hn -> hn (via h/agg scratch) -----
  run_gat_layer(hn, ei, W2, as2, ad2, b2, g2, be2,
                Wt, h, al, ar, mkey, ssum, pbuf, agg, stats, hn, stream);

  // ----- mean pool + FC -----
  (void)hipMemsetAsync(pooled, 0, (size_t)NGRAPH * HC * sizeof(float), stream);
  (void)hipMemsetAsync(cnt, 0, (size_t)NGRAPH * sizeof(float), stream);
  {
    const long long t = (long long)N_NODES * HC;
    pool_k<<<(unsigned)((t + 255) / 256), 256, 0, stream>>>(hn, batch, pooled);
  }
  cnt_k<<<(N_NODES + 255) / 256, 256, 0, stream>>>(batch, cnt);
  fc_k<<<(NGRAPH * NCLS + 255) / 256, 256, 0, stream>>>(pooled, cnt, fcW, fcb, out);
}
